// CPCircuitLayer_57483842289816
// MI455X (gfx1250) — compile-verified, matrix operand order checked by reading, not run on hardware
//
#include <hip/hip_runtime.h>

// CP circuit layer on gfx1250 (wave32, WMMA). B=4, S=1024, H=1024, R=32.
//   fused stage1+2 (one kernel, concurrent L2-sharing readers of hidden):
//     seq_h[b,s,r] = (sum_h hidden[b,s,h]*seq_W[r,h]) * cpw[r]   (f16 ws)
//     hid_h[b,h,r] =  sum_s hidden[b,s,h]*hid_W[r,s]             (f16 ws)
//   stage3: out[b,s,h] = sum_r seq_h[b,s,r]*hid_h[b,h,r]         (f32)
// Roofline: ~0.8 GFLOP vs ~32 MB HBM traffic (hidden read + out write; the
// second hidden reader and all ws traffic hit the 192 MB L2) -> memory bound,
// ~1.5 us floor at 23.3 TB/s. WMMA f16 with f32 accumulate (K=32 per v_wmma)
// covers all contractions; inputs are O(1) so f16 operand rounding is benign.

typedef __attribute__((ext_vector_type(16))) _Float16 v16h;
typedef __attribute__((ext_vector_type(8)))  float    v8f;

#define BB 4
#define SS 1024
#define HH 1024
#define RR 32
#define KCHUNK 256   // per-wave K range (4-way split-K)

// K index held by (element j, lane-half) in 16-bit A/B fragments (ISA 7.12.2):
//   lanes 0-15 : j=0..7 -> K=j,   j=8..15 -> K=j+8
//   lanes 16-31: j=0..7 -> K=j+8, j=8..15 -> K=j+16
__device__ __forceinline__ int kmap(int j, int half) {
  return j + 8 * half + ((j >= 8) ? 8 : 0);
}

__device__ __forceinline__ v16h load_row_frag_f32(const float* __restrict__ p) {
  // p = row base + 8*half; lane's two contiguous K-chunks at [0..7], [16..23]
  v16h r;
#pragma unroll
  for (int j = 0; j < 8; ++j) {
    r[j]     = (_Float16)p[j];
    r[j + 8] = (_Float16)p[16 + j];
  }
  return r;
}

__device__ __forceinline__ v8f wmma16(v16h a, v16h b, v8f c) {
  return __builtin_amdgcn_wmma_f32_16x16x32_f16(false, a, false, b, (short)0, c,
                                                false, false);
}

// ---------------- Fused stage 1 + 2: 256 blocks x 256 threads ----------------
// blocks [0,128): seq path, blocks [128,256): hid path. Each block owns 2
// output tiles (16 rows x 32 r); each tile is reduced by 4 waves over K/4.
// K-loops run over a compile-time-constant range so they stay wave-uniform
// (no EXEC masking around the WMMAs).
__global__ void __launch_bounds__(256)
emb_kernel(const float* __restrict__ hidden,   // (B*S, H)
           const float* __restrict__ seqW,     // (R, H)
           const float* __restrict__ hidW,     // (R, S)
           const float* __restrict__ cpw,      // (R)
           _Float16* __restrict__ seq_h,       // (B*S, R) f16
           _Float16* __restrict__ hid_h)       // (B*H, R) f16
{
  __shared__ float red[8][32][17];    // per-wave C spill (c0 | c1), padded
  __shared__ float tileS[8][32][17];  // hid path: wave-private 32(s)x16(h) stage

  const int lane = threadIdx.x & 31;
  const int widx = threadIdx.x >> 5;
  const int g    = widx >> 2;            // tile group in block (0..1)
  const int kw   = widx & 3;             // K-chunk (0..3)
  const int half = lane >> 4;
  const int m    = lane & 15;
  const int kbeg = kw * KCHUNK;

  v8f c0 = {}; v8f c1 = {};
  const bool isSeq = (blockIdx.x < 128);

  if (isSeq) {
    const int t    = blockIdx.x * 2 + g;           // 0..255
    const int row0 = t * 16;
    // fold kbeg + 8*half into the base pointers; loop bound is constant
    const float* aBase  = hidden + (size_t)(row0 + m) * HH + kbeg + 8 * half;
    const float* b0Base = seqW + (size_t)m * HH + kbeg + 8 * half;        // r=m
    const float* b1Base = seqW + (size_t)(m + 16) * HH + kbeg + 8 * half; // r=m+16
    for (int kk = 0; kk < KCHUNK; kk += 32) {
      v16h a  = load_row_frag_f32(aBase + kk);
      v16h b0 = load_row_frag_f32(b0Base + kk);
      v16h b1 = load_row_frag_f32(b1Base + kk);
      c0 = wmma16(a, b0, c0);
      c1 = wmma16(a, b1, c1);
    }
  } else {
    const int t  = (blockIdx.x - 128) * 2 + g;     // 0..255
    const int b  = t >> 6;
    const int h0 = (t & 63) * 16;
    const float* hBase  = hidden + ((size_t)b * SS + kbeg) * HH + h0 + m; // +s*HH
    const float* b0Base = hidW + (size_t)m * SS + kbeg + 8 * half;
    const float* b1Base = hidW + (size_t)(m + 16) * SS + kbeg + 8 * half;
    for (int kk = 0; kk < KCHUNK; kk += 32) {
      // wave-private transpose staging: tileS[s][h]; same-wave ds ordering
      // is enforced by DScnt waits, no block barrier needed.
#pragma unroll
      for (int ss = 0; ss < 16; ++ss) {
        const int s = ss * 2 + half;   // half-waves cover 16 consecutive h
        tileS[widx][s][m] = hBase[(size_t)(kk + s) * HH];
      }
      v16h a;
#pragma unroll
      for (int j = 0; j < 8; ++j) {
        a[j]     = (_Float16)tileS[widx][kmap(j, half)][m];
        a[j + 8] = (_Float16)tileS[widx][kmap(j + 8, half)][m];
      }
      v16h b0 = load_row_frag_f32(b0Base + kk);
      v16h b1 = load_row_frag_f32(b1Base + kk);
      c0 = wmma16(a, b0, c0);
      c1 = wmma16(a, b1, c1);
    }
  }

  // spill partial C fragments, reduce the 4 K-chunks across waves
#pragma unroll
  for (int i = 0; i < 8; ++i) {
    red[widx][lane][i]     = c0[i];
    red[widx][lane][i + 8] = c1[i];
  }
  __syncthreads();

  if (kw == 0) {
    float s0[8], s1[8];
#pragma unroll
    for (int i = 0; i < 8; ++i) {
      s0[i] = red[g * 4 + 0][lane][i] + red[g * 4 + 1][lane][i] +
              red[g * 4 + 2][lane][i] + red[g * 4 + 3][lane][i];
      s1[i] = red[g * 4 + 0][lane][i + 8] + red[g * 4 + 1][lane][i + 8] +
              red[g * 4 + 2][lane][i + 8] + red[g * 4 + 3][lane][i + 8];
    }
    if (isSeq) {
      const int row0 = (blockIdx.x * 2 + g) * 16;
      const float w0 = cpw[m], w1 = cpw[m + 16];   // fold cp_weight
#pragma unroll
      for (int i = 0; i < 8; ++i) {
        const int row = row0 + i + 8 * half;       // C/D: M = i+8*half, N = m
        seq_h[(size_t)row * RR + m]      = (_Float16)(s0[i] * w0);
        seq_h[(size_t)row * RR + 16 + m] = (_Float16)(s1[i] * w1);
      }
    } else {
      const int t  = (blockIdx.x - 128) * 2 + g;
      const int b  = t >> 6;
      const int h0 = (t & 63) * 16;
#pragma unroll
      for (int i = 0; i < 8; ++i) {
        const int h = h0 + i + 8 * half;
        hid_h[((size_t)b * HH + h) * RR + m]      = (_Float16)s0[i];
        hid_h[((size_t)b * HH + h) * RR + 16 + m] = (_Float16)s1[i];
      }
    }
  }
}

// ------------- Stage 3: out[b] = seq_h[b] @ hid_h[b]^T  (K = R = 32) ---------
__global__ void __launch_bounds__(256)
cp_out_kernel(const _Float16* __restrict__ seq_h,  // (B*S, R)
              const _Float16* __restrict__ hid_h,  // (B*H, R)
              float* __restrict__ out)             // (B, S, H)
{
  const int lane = threadIdx.x & 31;
  const int widx = threadIdx.x >> 5;
  const int gw   = blockIdx.x * 8 + widx;   // 0..2047
  const int b    = gw >> 9;
  const int rem  = gw & 511;
  const int s0   = (rem >> 3) * 16;
  const int h0   = (rem & 7) * 128;         // 8 h-tiles per wave, A reused
  const int half = lane >> 4;
  const int m    = lane & 15;

  v16h a;
  const _Float16* aRow = seq_h + ((size_t)b * SS + s0 + m) * RR + 8 * half;
#pragma unroll
  for (int j = 0; j < 8; ++j) { a[j] = aRow[j]; a[j + 8] = aRow[16 + j]; }

#pragma unroll
  for (int ht = 0; ht < 8; ++ht) {
    v16h bf;   // B[k,n] = hid_h[h0+ht*16+n, k]
    const _Float16* bRow =
        hid_h + ((size_t)b * HH + h0 + ht * 16 + m) * RR + 8 * half;
#pragma unroll
    for (int j = 0; j < 8; ++j) { bf[j] = bRow[j]; bf[j + 8] = bRow[16 + j]; }

    v8f c = {};
    c = wmma16(a, bf, c);

#pragma unroll
    for (int i = 0; i < 8; ++i) {
      out[((size_t)b * SS + s0 + i + 8 * half) * HH + h0 + ht * 16 + m] = c[i];
    }
  }
}

extern "C" void kernel_launch(void* const* d_in, const int* in_sizes, int n_in,
                              void* d_out, int out_size, void* d_ws, size_t ws_size,
                              hipStream_t stream) {
  (void)in_sizes; (void)n_in; (void)out_size; (void)ws_size;
  const float* hidden = (const float*)d_in[0];
  // d_in[1] = all_indices: identity gather (n//H, n%H) -> not needed
  const float* seqW = (const float*)d_in[2];
  const float* hidW = (const float*)d_in[3];
  const float* cpw  = (const float*)d_in[4];
  float* out = (float*)d_out;

  _Float16* seq_h = (_Float16*)d_ws;                                     // 256 KB
  _Float16* hid_h = (_Float16*)((char*)d_ws + (size_t)BB * SS * RR * 2); // 256 KB

  emb_kernel<<<256, 256, 0, stream>>>(hidden, seqW, hidW, cpw, seq_h, hid_h); // 2048 waves
  cp_out_kernel<<<256, 256, 0, stream>>>(seq_h, hid_h, out);                  // 2048 waves
}
